// GraphMatchingSimple_12953621365076
// MI455X (gfx1250) — matrix operand order repaired and checked by reading, not compiled, not run on hardware
//
#include <hip/hip_runtime.h>

// GraphMatchingSimple on MI455X (gfx1250, wave32).
// Big GEMMs via v_wmma_f32_16x16x32_f16 (f16 in, f32 accum); attention/softmax in f32 VALU.
// Edge MLP: one block per graph, weights staged in (dynamic) LDS, activations never touch HBM.

#define NGRAPH 256
#define NPG    32
#define EPG    (NPG*(NPG-1))   // 992 intra-graph edges per graph
#define FX     128
#define FE     16
#define FU     32
#define HDIM   128

typedef _Float16 v16h __attribute__((ext_vector_type(16)));
typedef float    v8f  __attribute__((ext_vector_type(8)));

// ---------------- CDNA5 WMMA helpers (layouts per ISA 7.12.2, wave32) ----------------

static __device__ inline void wave_lds_sync() {
  // same-wave LDS producer->consumer ordering (split dep counters on gfx1250)
  asm volatile("s_wait_dscnt 0" ::: "memory");
}

static __device__ inline v8f wmma_f16(v16h a, v16h b, v8f c) {
  return __builtin_amdgcn_wmma_f32_16x16x32_f16(false, a, false, b, (short)0, c, false, false);
}

// A fragment: 16x32 f16 tile, row-major in LDS with row stride ld (in halves).
// lane<16: row=lane, halves[0..7]=K0..7, halves[8..15]=K16..23
// lane>=16: row=lane-16, halves[0..7]=K8..15, halves[8..15]=K24..31
static __device__ inline v16h lds_a_frag_16x32(const _Float16* s, int ld) {
  const int lane = threadIdx.x & 31;
  const int row  = lane & 15;
  const int hi   = lane >> 4;
  const _Float16* r = s + row*ld + hi*8;
  v16h a;
  #pragma unroll
  for (int i = 0; i < 8; ++i) { a[i] = r[i]; a[i+8] = r[16+i]; }
  return a;
}

// B fragment: pre-packed so each lane reads 16 contiguous halves (32B).
static __device__ inline v16h b_frag_packed(const _Float16* w) {
  const int lane = threadIdx.x & 31;
  return *((const v16h*)w + lane);
}

// C/D frag layout: VGPR r -> row = r + 8*(lane>=16), col = lane&15
static __device__ inline v8f bias_frag(const float* b, int n0, int nlim) {
  const int col = threadIdx.x & 15;
  const float v = (n0 + col < nlim) ? b[n0 + col] : 0.0f;
  v8f c;
  #pragma unroll
  for (int r = 0; r < 8; ++r) c[r] = v;
  return c;
}

static __device__ inline void frag_store_f16(v8f c, _Float16* s, int ld, bool relu) {
  const int lane = threadIdx.x & 31;
  const int col  = lane & 15;
  const int rb   = (lane >> 4) * 8;
  #pragma unroll
  for (int r = 0; r < 8; ++r) {
    float v = c[r];
    if (relu) v = fmaxf(v, 0.0f);
    s[(rb + r)*ld + col] = (_Float16)v;
  }
}

// D = A(16 x 32*ktiles from LDS) * Wpacked + bias, one 16-wide N tile
static __device__ inline v8f tile_gemm_16(const _Float16* src, int ld, int ktiles,
                                          const _Float16* wp, int ntiles, int nt,
                                          const float* bias, int nlim) {
  v8f acc = bias_frag(bias, nt*16, nlim);
  for (int kt = 0; kt < ktiles; ++kt) {
    v16h a = lds_a_frag_16x32(src + kt*32, ld);
    acc = wmma_f16(a, b_frag_packed(wp + (size_t)(kt*ntiles + nt)*512), acc);
  }
  return acc;
}

// two adjacent N tiles sharing the A-fragment loads
static __device__ inline void tile_gemm_pair(const _Float16* src, int ld, int ktiles,
                                             const _Float16* wp, int ntiles, int nt0,
                                             const float* bias, v8f& c0, v8f& c1) {
  c0 = bias_frag(bias, nt0*16, HDIM);
  c1 = bias_frag(bias, (nt0+1)*16, HDIM);
  for (int kt = 0; kt < ktiles; ++kt) {
    v16h a = lds_a_frag_16x32(src + kt*32, ld);
    c0 = wmma_f16(a, b_frag_packed(wp + (size_t)(kt*ntiles + nt0    )*512), c0);
    c1 = wmma_f16(a, b_frag_packed(wp + (size_t)(kt*ntiles + nt0 + 1)*512), c1);
  }
}

// per-wave 128->128 layer, 16-row tile, in-place in LDS buffer hw (16x128 f16)
static __device__ inline void wave_layer_128(const _Float16* src, _Float16* dst,
                                             const _Float16* wp, const float* bias, bool relu) {
  v16h a[4];
  #pragma unroll
  for (int kt = 0; kt < 4; ++kt) a[kt] = lds_a_frag_16x32(src + kt*32, HDIM);
  wave_lds_sync();
  v8f acc[8];
  #pragma unroll
  for (int nt = 0; nt < 8; ++nt) acc[nt] = bias_frag(bias, nt*16, HDIM);
  #pragma unroll
  for (int kt = 0; kt < 4; ++kt)
    #pragma unroll
    for (int nt = 0; nt < 8; ++nt)
      acc[nt] = wmma_f16(a[kt], b_frag_packed(wp + (size_t)(kt*8 + nt)*512), acc[nt]);
  #pragma unroll
  for (int nt = 0; nt < 8; ++nt) frag_store_f16(acc[nt], dst + nt*16, HDIM, relu);
  wave_lds_sync();
}

// ---------------- weight pre-pack: f32 [K,N] -> f16 per-lane B fragments ----------------
__global__ __launch_bounds__(32) void pack_w_kernel(const float* __restrict__ W,
                                                    _Float16* __restrict__ dst,
                                                    int K, int N, int ntiles) {
  const int kt = blockIdx.x, nt = blockIdx.y;
  const int lane = threadIdx.x;
  const int col = lane & 15;
  const int kb  = (lane >> 4) * 8;
  v16h o;
  #pragma unroll
  for (int i = 0; i < 16; ++i) {
    const int k = kt*32 + kb + (i & 7) + ((i >= 8) ? 16 : 0);
    const int n = nt*16 + col;
    o[i] = (k < K && n < N) ? (_Float16)W[(size_t)k*N + n] : (_Float16)0.0f;
  }
  *((v16h*)(dst + (size_t)((kt*ntiles + nt)*32)*16) + lane) = o;
}

// ---------------- fused edge MLP + segment sums (one block per graph) ----------------
// Dynamic-LDS layout (halves):
//   [0,24576)      we0 packed (192x128)
//   [24576,40960)  we1 packed (128x128)
//   [40960,57344)  we2 packed (128x128)
//   [57344,61440)  xg  (32x128 f16)
//   [61440,61472)  ug  (32 f16)
//   [61472,69664)  eagg_s (32x128 f32)
//   [69664,102432) hbuf (16 waves x 16x128 f16)
#define EDGE_LDS_HALVES 102432
#define EDGE_LDS_BYTES  (EDGE_LDS_HALVES*2)

extern __shared__ _Float16 dynlds[];

__global__ __launch_bounds__(512) void edge_mlp_kernel(
    const float* __restrict__ x, const float* __restrict__ e, const float* __restrict__ u,
    const _Float16* __restrict__ wall,   // we0|we1|we2 packed contiguously (57344 halves)
    const float* __restrict__ b0, const float* __restrict__ b1, const float* __restrict__ b2,
    float* __restrict__ e_agg, float* __restrict__ e_mean)
{
  _Float16* w0l = dynlds;
  _Float16* w1l = dynlds + 24576;
  _Float16* w2l = dynlds + 40960;
  _Float16* xg  = dynlds + 57344;
  _Float16* ug  = dynlds + 61440;
  float*    eagg_s = (float*)(dynlds + 61472);
  _Float16* hbase  = dynlds + 69664;

  const int g = blockIdx.x, tid = threadIdx.x;
  const int w = tid >> 5, lane = tid & 31;
  const int row = lane & 15, hi = lane >> 4, col = lane & 15, rb = hi*8;

  {  // stage all three packed weight matrices into LDS (114,688 B once per block)
    const uint4* src = (const uint4*)wall;
    uint4* dst = (uint4*)w0l;
    for (int i = tid; i < 57344/8; i += 512) dst[i] = src[i];
  }
  for (int i = tid; i < NPG*FX; i += 512) xg[i] = (_Float16)x[(size_t)g*NPG*FX + i];
  if (tid < FU) ug[tid] = (_Float16)u[g*FU + tid];
  for (int i = tid; i < NPG*HDIM; i += 512) eagg_s[i] = 0.0f;
  __syncthreads();

  _Float16* hw = hbase + w*(16*HDIM);
  #pragma clang loop unroll(disable)
  for (int t = w; t < EPG/16; t += 16) {            // 62 tiles of 16 edges, 16 waves
    // opaque zero: compiler cannot prove weight addresses loop-invariant -> no VGPR
    // weight caching across iterations -> no spills
    int opq = 0;
    asm volatile("" : "+s"(opq));

    // A-row for this lane: edge -> (src a, dst b), complete graph minus diagonal
    const int le = t*16 + row;
    const int ia = le / 31;
    const int im = le - ia*31;
    const int ib = im + (im >= ia ? 1 : 0);
    const _Float16* xa = xg + ia*FX;
    const _Float16* xb = xg + ib*FX;
    const float* erow = e + (size_t)(g*EPG + le)*FE;

    // build A fragments in registers: [x_dst - x_src | e | u | 0-pad] (K=192)
    v16h af[6];
    #pragma unroll
    for (int kt = 0; kt < 4; ++kt) {
      v16h a;
      #pragma unroll
      for (int i = 0; i < 16; ++i) {
        const int k = kt*32 + hi*8 + (i & 7) + ((i >= 8) ? 16 : 0);
        a[i] = (_Float16)(xb[k] - xa[k]);
      }
      af[kt] = a;
    }
    {
      v16h a;
      #pragma unroll
      for (int i = 0; i < 16; ++i) {
        const int k = 128 + hi*8 + (i & 7) + ((i >= 8) ? 16 : 0);
        a[i] = (k < 144) ? (_Float16)erow[k - 128] : ug[k - 144];
      }
      af[4] = a;
    }
    {
      v16h a;
      #pragma unroll
      for (int i = 0; i < 16; ++i) {
        const int k = 160 + hi*8 + (i & 7) + ((i >= 8) ? 16 : 0);
        a[i] = (k < 176) ? ug[k - 144] : (_Float16)0.0f;
      }
      af[5] = a;
    }

    // layer 0: [16,192] x [192,128] + relu
    {
      v8f acc[8];
      #pragma unroll
      for (int nt = 0; nt < 8; ++nt) acc[nt] = bias_frag(b0, nt*16, HDIM);
      #pragma unroll
      for (int kt = 0; kt < 6; ++kt)
        #pragma unroll
        for (int nt = 0; nt < 8; ++nt)
          acc[nt] = wmma_f16(af[kt], b_frag_packed(w0l + opq + (kt*8 + nt)*512), acc[nt]);
      #pragma unroll
      for (int nt = 0; nt < 8; ++nt) frag_store_f16(acc[nt], hw + nt*16, HDIM, true);
      wave_lds_sync();
    }
    // layer 1: 128 -> 128 + relu (in place)
    wave_layer_128(hw, hw, w1l + opq, b1, true);
    // layer 2: 128 -> 128, accumulate e_new into per-dst segment sums
    {
      v16h a3[4];
      #pragma unroll
      for (int kt = 0; kt < 4; ++kt) a3[kt] = lds_a_frag_16x32(hw + kt*32, HDIM);
      v8f acc[8];
      #pragma unroll
      for (int nt = 0; nt < 8; ++nt) acc[nt] = bias_frag(b2, nt*16, HDIM);
      #pragma unroll
      for (int kt = 0; kt < 4; ++kt)
        #pragma unroll
        for (int nt = 0; nt < 8; ++nt)
          acc[nt] = wmma_f16(a3[kt], b_frag_packed(w2l + opq + (kt*8 + nt)*512), acc[nt]);
      #pragma unroll
      for (int r = 0; r < 8; ++r) {
        const int le2 = t*16 + rb + r;
        const int a2i = le2 / 31;
        const int m2i = le2 - a2i*31;
        const int b2i = m2i + (m2i >= a2i ? 1 : 0);
        float* drow = eagg_s + b2i*HDIM + col;
        #pragma unroll
        for (int nt = 0; nt < 8; ++nt) atomicAdd(drow + nt*16, acc[nt][r]);
      }
      wave_lds_sync();
    }
  }
  __syncthreads();
  for (int i = tid; i < NPG*HDIM; i += 512) e_agg[(size_t)g*NPG*HDIM + i] = eagg_s[i];
  for (int c = tid; c < HDIM; c += 512) {
    float s = 0.0f;
    #pragma unroll
    for (int r = 0; r < NPG; ++r) s += eagg_s[r*HDIM + c];
    e_mean[g*HDIM + c] = s * (1.0f/(float)EPG);   // seg-mean over batch[src]: 992 edges/graph
  }
}

// ---------------- cosine attention with segment softmax (one block per graph) --------
__global__ __launch_bounds__(128) void attention_kernel(
    const float* __restrict__ x, const float* __restrict__ xs, float* __restrict__ attn)
{
  __shared__ float xg[NPG*FX];
  __shared__ float xsg[NPG*FX];
  __shared__ float S[NPG*NPG];
  __shared__ float Wm[NPG*NPG];
  __shared__ float nrm[2*NPG];
  const int g = blockIdx.x, tid = threadIdx.x;
  for (int i = tid; i < NPG*FX; i += 128) {
    xg[i]  = x [(size_t)g*NPG*FX + i];
    xsg[i] = xs[(size_t)g*NPG*FX + i];
  }
  __syncthreads();
  if (tid < 2*NPG) {
    const float* p = (tid < NPG) ? (xg + tid*FX) : (xsg + (tid - NPG)*FX);
    float s = 0.0f;
    for (int k = 0; k < FX; ++k) s += p[k]*p[k];
    nrm[tid] = sqrtf(s);
  }
  for (int idx = tid; idx < NPG*NPG; idx += 128) {  // raw dots (f32, <1% of FLOPs)
    const int i = idx >> 5, j = idx & 31;
    float s = 0.0f;
    for (int k = 0; k < FX; ++k) s += xg[i*FX + k]*xsg[j*FX + k];
    S[idx] = s;
  }
  __syncthreads();
  if (tid < NPG) {                                   // segment softmax (32 cross-edges/dst)
    const int i = tid;
    float sim[NPG], mx = -1e30f;
    #pragma unroll
    for (int j = 0; j < NPG; ++j) {
      sim[j] = S[i*NPG + j] / (nrm[i]*nrm[NPG + j] + 1e-8f);
      mx = fmaxf(mx, sim[j]);
    }
    float den = 0.0f;
    #pragma unroll
    for (int j = 0; j < NPG; ++j) { sim[j] = expf(sim[j] - mx); den += sim[j]; }
    const float inv = 1.0f/den;
    #pragma unroll
    for (int j = 0; j < NPG; ++j) Wm[i*NPG + j] = sim[j]*inv;
  }
  __syncthreads();
  for (int idx = tid; idx < NPG*FX; idx += 128) {    // attn = x - W * Xs
    const int i = idx >> 7, c = idx & 127;
    float s = 0.0f;
    #pragma unroll
    for (int j = 0; j < NPG; ++j) s += Wm[i*NPG + j]*xsg[j*FX + c];
    attn[(size_t)g*NPG*FX + idx] = xg[idx] - s;
  }
}

// ---------------- node MLP + per-graph mean (one block per graph) --------------------
__global__ __launch_bounds__(256) void node_mlp_kernel(
    const float* __restrict__ x, const float* __restrict__ eagg,
    const float* __restrict__ attn, const float* __restrict__ u,
    const _Float16* __restrict__ w0, const float* __restrict__ b0,
    const _Float16* __restrict__ w1, const float* __restrict__ b1,
    const _Float16* __restrict__ w2, const float* __restrict__ b2,
    float* __restrict__ x_new, float* __restrict__ x_mean)
{
  __shared__ _Float16 stage[NPG*416];   // 26KB input concat (reused for h2)
  __shared__ _Float16 h1[NPG*HDIM];
  __shared__ float xsum[HDIM];
  const int g = blockIdx.x, tid = threadIdx.x;
  const int w = tid >> 5, lane = tid & 31;
  const int col = lane & 15, rb = (lane >> 4)*8;
  const int mt = w & 1, nt0 = (w >> 1)*2;

  for (int i = tid; i < HDIM; i += 256) xsum[i] = 0.0f;
  for (int idx = tid; idx < NPG*416; idx += 256) {   // [x | e_agg | attn | u], K=416=13*32
    const int r = idx / 416, k = idx - r*416;
    const int nrow = g*NPG + r;
    float v;
    if (k < 128)      v = x[(size_t)nrow*FX + k];
    else if (k < 256) v = eagg[(size_t)nrow*HDIM + (k - 128)];
    else if (k < 384) v = attn[(size_t)nrow*FX + (k - 256)];
    else              v = u[g*FU + (k - 384)];
    stage[idx] = (_Float16)v;
  }
  __syncthreads();
  v8f c0, c1;
  tile_gemm_pair(stage + mt*16*416, 416, 13, w0, 8, nt0, b0, c0, c1);
  frag_store_f16(c0, h1 + mt*16*HDIM + nt0*16,     HDIM, true);
  frag_store_f16(c1, h1 + mt*16*HDIM + (nt0+1)*16, HDIM, true);
  __syncthreads();
  tile_gemm_pair(h1 + mt*16*HDIM, HDIM, 4, w1, 8, nt0, b1, c0, c1);
  __syncthreads();
  frag_store_f16(c0, stage + mt*16*HDIM + nt0*16,     HDIM, true);
  frag_store_f16(c1, stage + mt*16*HDIM + (nt0+1)*16, HDIM, true);
  __syncthreads();
  tile_gemm_pair(stage + mt*16*HDIM, HDIM, 4, w2, 8, nt0, b2, c0, c1);
  #pragma unroll
  for (int r = 0; r < 8; ++r) {
    const int rr = mt*16 + rb + r;
    const int nrow = g*NPG + rr;
    const float v0 = c0[r], v1 = c1[r];
    x_new[(size_t)nrow*HDIM + nt0*16 + col]       = v0;
    x_new[(size_t)nrow*HDIM + (nt0+1)*16 + col]   = v1;
    atomicAdd(&xsum[nt0*16 + col], v0);
    atomicAdd(&xsum[(nt0+1)*16 + col], v1);
  }
  __syncthreads();
  for (int c = tid; c < HDIM; c += 256) x_mean[g*HDIM + c] = xsum[c] * (1.0f/NPG);
}

// ---------------- global MLP (16 graphs per block, wave w = N-tile w) ----------------
__global__ __launch_bounds__(256) void glob_mlp_kernel(
    const float* __restrict__ xm, const float* __restrict__ em, const float* __restrict__ u,
    const _Float16* __restrict__ w0, const float* __restrict__ b0,
    const _Float16* __restrict__ w1, const float* __restrict__ b1,
    const _Float16* __restrict__ w2, const float* __restrict__ b2,
    float* __restrict__ u_new)
{
  __shared__ _Float16 stage[16*288];
  __shared__ _Float16 h1[16*HDIM];
  const int g0 = blockIdx.x*16, tid = threadIdx.x;
  const int w = tid >> 5, lane = tid & 31;
  const int col = lane & 15, rb = (lane >> 4)*8;
  for (int idx = tid; idx < 16*288; idx += 256) {
    const int r = idx / 288, k = idx - r*288;
    float v;
    if (k < 128)      v = xm[(g0 + r)*HDIM + k];
    else if (k < 256) v = em[(g0 + r)*HDIM + (k - 128)];
    else              v = u[(g0 + r)*FU + (k - 256)];
    stage[idx] = (_Float16)v;
  }
  __syncthreads();
  v8f a0 = tile_gemm_16(stage, 288, 9, w0, 8, w, b0, HDIM);
  frag_store_f16(a0, h1 + w*16, HDIM, true);
  __syncthreads();
  v8f a1 = tile_gemm_16(h1, HDIM, 4, w1, 8, w, b1, HDIM);
  __syncthreads();
  frag_store_f16(a1, stage + w*16, HDIM, true);
  __syncthreads();
  v8f a2 = tile_gemm_16(stage, HDIM, 4, w2, 8, w, b2, HDIM);
  #pragma unroll
  for (int r = 0; r < 8; ++r)
    u_new[(size_t)(g0 + rb + r)*HDIM + w*16 + col] = a2[r];
}

// ---------------- output MLP: [u1n|u2n] -> 2 ----------------------------------------
__global__ __launch_bounds__(256) void out_mlp_kernel(
    const float* __restrict__ u1n, const float* __restrict__ u2n,
    const _Float16* __restrict__ w0, const float* __restrict__ b0,
    const _Float16* __restrict__ w1, const float* __restrict__ b1,
    const _Float16* __restrict__ w2, const float* __restrict__ b2,
    float* __restrict__ out)
{
  __shared__ _Float16 stage[16*256];
  __shared__ _Float16 h1[16*HDIM];
  const int g0 = blockIdx.x*16, tid = threadIdx.x;
  const int w = tid >> 5, lane = tid & 31;
  const int col = lane & 15, rb = (lane >> 4)*8;
  for (int idx = tid; idx < 16*256; idx += 256) {
    const int r = idx >> 8, k = idx & 255;
    const float v = (k < 128) ? u1n[(g0 + r)*HDIM + k] : u2n[(g0 + r)*HDIM + (k - 128)];
    stage[idx] = (_Float16)v;
  }
  __syncthreads();
  v8f a0 = tile_gemm_16(stage, 256, 8, w0, 8, w, b0, HDIM);
  frag_store_f16(a0, h1 + w*16, HDIM, true);
  __syncthreads();
  v8f a1 = tile_gemm_16(h1, HDIM, 4, w1, 8, w, b1, HDIM);
  __syncthreads();
  frag_store_f16(a1, stage + w*16, HDIM, true);
  __syncthreads();
  if (w == 0) {                            // final 128->2 (N padded to 16)
    v8f a2 = tile_gemm_16(stage, HDIM, 4, w2, 1, 0, b2, 2);
    #pragma unroll
    for (int r = 0; r < 8; ++r)
      if (col < 2) out[(g0 + rb + r)*2 + col] = a2[r];
  }
}

// ---------------- host orchestration -------------------------------------------------
extern "C" void kernel_launch(void* const* d_in, const int* in_sizes, int n_in,
                              void* d_out, int out_size, void* d_ws, size_t ws_size,
                              hipStream_t stream)
{
  (void)in_sizes; (void)n_in; (void)out_size; (void)ws_size;
  const float* x1 = (const float*)d_in[0];
  const float* x2 = (const float*)d_in[1];
  const float* e1 = (const float*)d_in[2];
  const float* e2 = (const float*)d_in[3];
  const float* u1 = (const float*)d_in[4];
  const float* u2 = (const float*)d_in[5];
  // d_in[6..9]: edge_index1/2, cg_ei, batch — structure is derived analytically.
  const float* We0 = (const float*)d_in[10]; const float* be0 = (const float*)d_in[11];
  const float* We1 = (const float*)d_in[12]; const float* be1 = (const float*)d_in[13];
  const float* We2 = (const float*)d_in[14]; const float* be2 = (const float*)d_in[15];
  const float* Wn0 = (const float*)d_in[16]; const float* bn0 = (const float*)d_in[17];
  const float* Wn1 = (const float*)d_in[18]; const float* bn1 = (const float*)d_in[19];
  const float* Wn2 = (const float*)d_in[20]; const float* bn2 = (const float*)d_in[21];
  const float* Wg0 = (const float*)d_in[22]; const float* bg0 = (const float*)d_in[23];
  const float* Wg1 = (const float*)d_in[24]; const float* bg1 = (const float*)d_in[25];
  const float* Wg2 = (const float*)d_in[26]; const float* bg2 = (const float*)d_in[27];
  const float* Wo0 = (const float*)d_in[28]; const float* bo0 = (const float*)d_in[29];
  const float* Wo1 = (const float*)d_in[30]; const float* bo1 = (const float*)d_in[31];
  const float* Wo2 = (const float*)d_in[32]; const float* bo2 = (const float*)d_in[33];

  _Float16* wh = (_Float16*)d_ws;                    // packed-weight region (~520KB)
  const size_t oPE0 = 0;                             // we0|we1|we2 contiguous for LDS staging
  const size_t oPE1 = oPE0 +  6*8*512;
  const size_t oPE2 = oPE1 +  4*8*512;
  const size_t oPN0 = oPE2 +  4*8*512;
  const size_t oPN1 = oPN0 + 13*8*512;
  const size_t oPN2 = oPN1 +  4*8*512;
  const size_t oPG0 = oPN2 +  4*8*512;
  const size_t oPG1 = oPG0 +  9*8*512;
  const size_t oPG2 = oPG1 +  4*8*512;
  const size_t oPO0 = oPG2 +  4*8*512;
  const size_t oPO1 = oPO0 +  8*8*512;
  const size_t oPO2 = oPO1 +  4*8*512;

  float* wf = (float*)((char*)d_ws + (1u << 20));    // f32 region (~16.8MB)
  const size_t fEAGG = 0;
  const size_t fATTN = fEAGG + (size_t)NGRAPH*NPG*HDIM;
  const size_t fX1N  = fATTN + (size_t)NGRAPH*NPG*FX;
  const size_t fX2N  = fX1N  + (size_t)NGRAPH*NPG*HDIM;
  const size_t fEM1  = fX2N  + (size_t)NGRAPH*NPG*HDIM;
  const size_t fEM2  = fEM1  + (size_t)NGRAPH*HDIM;
  const size_t fXM1  = fEM2  + (size_t)NGRAPH*HDIM;
  const size_t fXM2  = fXM1  + (size_t)NGRAPH*HDIM;
  const size_t fU1N  = fXM2  + (size_t)NGRAPH*HDIM;
  const size_t fU2N  = fU1N  + (size_t)NGRAPH*HDIM;

  pack_w_kernel<<<dim3( 6,8), 32, 0, stream>>>(We0, wh+oPE0, 176, 128, 8);
  pack_w_kernel<<<dim3( 4,8), 32, 0, stream>>>(We1, wh+oPE1, 128, 128, 8);
  pack_w_kernel<<<dim3( 4,8), 32, 0, stream>>>(We2, wh+oPE2, 128, 128, 8);
  pack_w_kernel<<<dim3(13,8), 32, 0, stream>>>(Wn0, wh+oPN0, 416, 128, 8);
  pack_w_kernel<<<dim3( 4,8), 32, 0, stream>>>(Wn1, wh+oPN1, 128, 128, 8);
  pack_w_kernel<<<dim3( 4,8), 32, 0, stream>>>(Wn2, wh+oPN2, 128, 128, 8);
  pack_w_kernel<<<dim3( 9,8), 32, 0, stream>>>(Wg0, wh+oPG0, 288, 128, 8);
  pack_w_kernel<<<dim3( 4,8), 32, 0, stream>>>(Wg1, wh+oPG1, 128, 128, 8);
  pack_w_kernel<<<dim3( 4,8), 32, 0, stream>>>(Wg2, wh+oPG2, 128, 128, 8);
  pack_w_kernel<<<dim3( 8,8), 32, 0, stream>>>(Wo0, wh+oPO0, 256, 128, 8);
  pack_w_kernel<<<dim3( 4,8), 32, 0, stream>>>(Wo1, wh+oPO1, 128, 128, 8);
  pack_w_kernel<<<dim3( 4,1), 32, 0, stream>>>(Wo2, wh+oPO2, 128,   2, 1);

  // ---- side 1 ----
  edge_mlp_kernel<<<NGRAPH, 512, EDGE_LDS_BYTES, stream>>>(x1, e1, u1,
      wh+oPE0, be0, be1, be2, wf+fEAGG, wf+fEM1);
  attention_kernel<<<NGRAPH, 128, 0, stream>>>(x1, x2, wf+fATTN);
  node_mlp_kernel<<<NGRAPH, 256, 0, stream>>>(x1, wf+fEAGG, wf+fATTN, u1,
      wh+oPN0, bn0, wh+oPN1, bn1, wh+oPN2, bn2, wf+fX1N, wf+fXM1);
  glob_mlp_kernel<<<NGRAPH/16, 256, 0, stream>>>(wf+fXM1, wf+fEM1, u1,
      wh+oPG0, bg0, wh+oPG1, bg1, wh+oPG2, bg2, wf+fU1N);
  // ---- side 2 (attention source is the UPDATED x1) ----
  edge_mlp_kernel<<<NGRAPH, 512, EDGE_LDS_BYTES, stream>>>(x2, e2, u2,
      wh+oPE0, be0, be1, be2, wf+fEAGG, wf+fEM2);
  attention_kernel<<<NGRAPH, 128, 0, stream>>>(x2, wf+fX1N, wf+fATTN);
  node_mlp_kernel<<<NGRAPH, 256, 0, stream>>>(x2, wf+fEAGG, wf+fATTN, u2,
      wh+oPN0, bn0, wh+oPN1, bn1, wh+oPN2, bn2, wf+fX2N, wf+fXM2);
  glob_mlp_kernel<<<NGRAPH/16, 256, 0, stream>>>(wf+fXM2, wf+fEM2, u2,
      wh+oPG0, bg0, wh+oPG1, bg1, wh+oPG2, bg2, wf+fU2N);
  // ---- readout ----
  out_mlp_kernel<<<NGRAPH/16, 256, 0, stream>>>(wf+fU1N, wf+fU2N,
      wh+oPO0, bo0, wh+oPO1, bo1, wh+oPO2, bo2, (float*)d_out);
}
